// Attention_25383256719981
// MI455X (gfx1250) — matrix operand order repaired and checked by reading, not compiled, not run on hardware
//
#include <hip/hip_runtime.h>
#include <hip/hip_bf16.h>

typedef __attribute__((ext_vector_type(16))) _Float16 v16h;
typedef __attribute__((ext_vector_type(8)))  float    v8f;

// Problem constants (match reference)
#define KB 2
#define KN 2048
#define KD 1024
#define KH 16
#define KDH 64
#define QK_SCALE 0.125f      // 64^-0.5
#define NEGINF (-1.0e30f)

// ---------------------------------------------------------------------------
// CDNA5 async copy global->LDS (ASYNCcnt-tracked, bypasses VGPRs).
// GVS addressing: mem = SADDR(sgpr64) + voff(vgpr32) ; LDS dest = vgpr32 offset.
// ---------------------------------------------------------------------------
__device__ __forceinline__ unsigned lds_off(const void* p) {
  // generic pointers to LDS carry the byte offset in the low 32 bits
  return (unsigned)(unsigned long long)p;
}
__device__ __forceinline__ void async_b128(unsigned lds, unsigned voff,
                                           const void* sbase) {
  asm volatile("global_load_async_to_lds_b128 %0, %1, %2"
               :: "v"(lds), "v"(voff), "s"(sbase) : "memory");
}
#define WAIT_ASYNC(n) asm volatile("s_wait_asynccnt " #n ::: "memory")

// ---------------------------------------------------------------------------
// WMMA helpers: operand loaders follow the CDNA5 ISA 16-bit VGPR layouts.
// ---------------------------------------------------------------------------
__device__ __forceinline__ v8f wmma_f16(v16h a, v16h b, v8f c) {
  return __builtin_amdgcn_wmma_f32_16x16x32_f16(false, a, false, b,
                                                (short)0, c, false, false);
}

// A operand (16x32, MxK). base points at row-major halves, row stride ld.
// lane<16: M=lane, K = {0..7,16..23}; lane>=16: M=lane-16, K = {8..15,24..31}.
__device__ __forceinline__ v16h load_A(const _Float16* base, int ld, int k0, int lane) {
  union { v16h v; unsigned u[8]; } r;
  const _Float16* row = base + (size_t)(lane & 15) * ld + k0 + ((lane < 16) ? 0 : 8);
#pragma unroll
  for (int i = 0; i < 8; ++i) {
    int k = ((i >> 2) << 4) + ((i & 3) << 1);   // group*16 + pair*2
    r.u[i] = *(const unsigned*)(row + k);
  }
  return r.v;
}

// B operand (32x16, KxN) read from a buffer laid out [n][k] (row n, K contig).
__device__ __forceinline__ v16h load_B(const _Float16* base, int ld, int k0, int lane) {
  union { v16h v; unsigned u[8]; } r;
  const _Float16* row = base + (size_t)(lane & 15) * ld + k0 + ((lane < 16) ? 0 : 8);
#pragma unroll
  for (int i = 0; i < 8; ++i) r.u[i] = *(const unsigned*)(row + ((i >> 2) << 4) + ((i & 3) << 1));
  return r.v;
}

// ---------------------------------------------------------------------------
// f32 -> f16 conversion
// ---------------------------------------------------------------------------
__global__ __launch_bounds__(256) void cvt_f32_f16(const float* __restrict__ in,
                                                   _Float16* __restrict__ out, int n) {
  int i = blockIdx.x * 256 + threadIdx.x;
  if (i < n) out[i] = (_Float16)in[i];
}

// ---------------------------------------------------------------------------
// WMMA GEMM: C(M,Nc) = A(M,K) * W(Nc,K)^T [+ bias]
// 256 threads = 8 waves (2x4); block tile 64(M) x 256(N); K step 32.
// Double-buffered LDS fed by async copies.
// ---------------------------------------------------------------------------
__device__ __forceinline__ void gemm_issue_stage(
    const _Float16* A, const _Float16* W, int Kd, int mb, int nb, int k0,
    _Float16* AsBuf, _Float16* WsBuf, int tid) {
  {  // A tile: 64 rows x 64B -> 256 x b128 (1 per thread)
    int row = tid >> 2, cu = tid & 3;
    async_b128(lds_off(AsBuf + row * 32 + cu * 8),
               (unsigned)((((size_t)(mb + row)) * Kd + k0 + cu * 8) * 2), A);
  }
#pragma unroll
  for (int j = 0; j < 4; ++j) {  // W tile: 256 rows x 64B -> 1024 x b128
    int c = tid + 256 * j;
    int row = c >> 2, cu = c & 3;
    async_b128(lds_off(WsBuf + row * 32 + cu * 8),
               (unsigned)((((size_t)(nb + row)) * Kd + k0 + cu * 8) * 2), W);
  }
}

template <bool OUT_F16, bool ADD_BIAS>
__global__ __launch_bounds__(256) void gemm_wmma(
    const _Float16* __restrict__ A, const _Float16* __restrict__ W,
    const float* __restrict__ bias, float* __restrict__ Cf,
    _Float16* __restrict__ Ch, int M, int Nc, int Kd) {
  __shared__ __align__(16) _Float16 As[2][64 * 32];
  __shared__ __align__(16) _Float16 Ws[2][256 * 32];
  const int tid  = threadIdx.x;
  const int lane = tid & 31;
  const int wid  = tid >> 5;
  const int wy = wid >> 2, wx = wid & 3;          // 2 x 4 wave grid
  const int mb = blockIdx.x * 64, nb = blockIdx.y * 256;

  v8f acc[2][4] = {};

  const int nk = Kd >> 5;
  gemm_issue_stage(A, W, Kd, mb, nb, 0, As[0], Ws[0], tid);

  for (int ks = 0; ks < nk; ++ks) {
    __syncthreads();                       // prior readers of the write buffer done
    if (ks + 1 < nk) {
      gemm_issue_stage(A, W, Kd, mb, nb, (ks + 1) << 5,
                       As[(ks + 1) & 1], Ws[(ks + 1) & 1], tid);
      WAIT_ASYNC(5);                       // current stage landed; next in flight
    } else {
      WAIT_ASYNC(0);
    }
    __syncthreads();

    const _Float16* Asb = As[ks & 1];
    const _Float16* Wsb = Ws[ks & 1];
    v16h a0 = load_A(Asb + (wy * 32 + 0) * 32, 32, 0, lane);
    v16h a1 = load_A(Asb + (wy * 32 + 16) * 32, 32, 0, lane);
#pragma unroll
    for (int t = 0; t < 4; ++t) {
      v16h b = load_B(Wsb + (wx * 64 + t * 16) * 32, 32, 0, lane);
      acc[0][t] = wmma_f16(a0, b, acc[0][t]);
      acc[1][t] = wmma_f16(a1, b, acc[1][t]);
    }
  }

  const int n0 = lane & 15;
#pragma unroll
  for (int sm = 0; sm < 2; ++sm)
#pragma unroll
    for (int t = 0; t < 4; ++t)
#pragma unroll
      for (int r = 0; r < 8; ++r) {
        int m = mb + wy * 32 + sm * 16 + ((lane < 16) ? r : r + 8);
        int n = nb + wx * 64 + t * 16 + n0;
        float v = acc[sm][t][r];
        if (ADD_BIAS) v += bias[n];
        if (OUT_F16) Ch[(size_t)m * Nc + n] = (_Float16)v;
        else         Cf[(size_t)m * Nc + n] = v;
      }
}

// ---------------------------------------------------------------------------
// RoPE + head split. qkv: (B*N, 3*H*DH) f16.
// q,k -> (B,H,N,DH) f16 ; v -> TRANSPOSED (B,H,DH,N) f16 so flash-attn can
// stage V^T tiles with linear async copies. q pre-scaled; rotary on q,k,v.
// ---------------------------------------------------------------------------
__global__ __launch_bounds__(256) void rope_split(
    const _Float16* __restrict__ qkv, const float* __restrict__ freqs,
    _Float16* __restrict__ q, _Float16* __restrict__ k,
    _Float16* __restrict__ vT) {
  int tid = blockIdx.x * 256 + threadIdx.x;
  const int total = KB * KN * 3 * KH * (KDH / 2);
  if (tid >= total) return;
  int flat = tid * 2;
  int row = flat / (3 * KH * KDH);           // b*N + n
  int c   = flat % (3 * KH * KDH);
  int b = row / KN, n = row % KN;
  int which = c / (KH * KDH);
  int rem   = c % (KH * KDH);
  int h = rem / KDH, dh = rem % KDH;

  float t0 = (float)qkv[(size_t)row * (3 * KH * KDH) + c];
  float t1 = (float)qkv[(size_t)row * (3 * KH * KDH) + c + 1];
  float f0 = freqs[n * KDH + dh];
  float f1 = freqs[n * KDH + dh + 1];
  // rot[2i] = -t[2i+1], rot[2i+1] = t[2i]
  float o0 = t0 * __cosf(f0) - t1 * __sinf(f0);
  float o1 = t1 * __cosf(f1) + t0 * __sinf(f1);

  if (which == 2) {
    size_t idx = (((size_t)b * KH + h) * KDH + dh) * KN + n;
    vT[idx]      = (_Float16)o0;
    vT[idx + KN] = (_Float16)o1;           // dh+1 row
  } else {
    if (which == 0) { o0 *= QK_SCALE; o1 *= QK_SCALE; }
    _Float16* dst = (which == 0) ? q : k;
    size_t idx = (((size_t)b * KH + h) * KN + n) * KDH + dh;
    dst[idx]     = (_Float16)o0;
    dst[idx + 1] = (_Float16)o1;
  }
}

// ---------------------------------------------------------------------------
// Flash attention (causal + key mask). grid = (N/64, H, B), block = 128 (4 waves).
// Each wave owns 16 q rows; block shares double-buffered 32-key K / V^T tiles
// staged with async copies. Row sums of P computed with a ones-matrix WMMA.
// ---------------------------------------------------------------------------
__device__ __forceinline__ void attn_issue_stage(
    const _Float16* Kbh, const _Float16* VTbh, int jb,
    _Float16* KsBuf, _Float16* VTsBuf, int tid) {
#pragma unroll
  for (int j = 0; j < 2; ++j) {  // K tile: 32 keys x 128B = 256 x b128
    int c = tid + 128 * j;
    int row = c >> 3, cu = c & 7;
    async_b128(lds_off(KsBuf + row * 64 + cu * 8),
               (unsigned)((((size_t)(jb + row)) * KDH + cu * 8) * 2), Kbh);
  }
#pragma unroll
  for (int j = 0; j < 2; ++j) {  // V^T tile: 64 dh-rows x 64B = 256 x b128
    int c = tid + 128 * j;
    int row = c >> 2, cu = c & 3;
    async_b128(lds_off(VTsBuf + row * 32 + cu * 8),
               (unsigned)((((size_t)row) * KN + jb + cu * 8) * 2), VTbh);
  }
}

__global__ __launch_bounds__(128) void flash_attn(
    const _Float16* __restrict__ Q, const _Float16* __restrict__ K,
    const _Float16* __restrict__ VT, const unsigned char* __restrict__ mask,
    _Float16* __restrict__ Out) {
  __shared__ __align__(16) _Float16 Ks[2][32 * 64];     // [key][dh]
  __shared__ __align__(16) _Float16 VTs[2][64 * 32];    // [dh][key]
  __shared__ __align__(16) _Float16 Ps[4][16 * 32];     // per-wave P [m][k]

  const int tid  = threadIdx.x;
  const int lane = tid & 31;
  const int wid  = tid >> 5;
  const int n0   = lane & 15;
  const int qb   = blockIdx.x * 64;
  const int h = blockIdx.y, b = blockIdx.z;
  const int qbase = qb + wid * 16;

  const _Float16* Qbh  = Q  + (((size_t)b * KH + h) * KN) * KDH;
  const _Float16* Kbh  = K  + (((size_t)b * KH + h) * KN) * KDH;
  const _Float16* VTbh = VT + (((size_t)b * KH + h) * KDH) * KN;
  const unsigned char* maskb = mask + (size_t)b * KN;

  // Q tile (16x64) as two A operands (K = 0..31, 32..63); q pre-scaled.
  v16h qa0 = load_A(Qbh + (size_t)qbase * KDH, KDH, 0, lane);
  v16h qa1 = load_A(Qbh + (size_t)qbase * KDH, KDH, 32, lane);

  // all-ones B operand for row-sum WMMA
  v16h ones;
#pragma unroll
  for (int i = 0; i < 16; ++i) ones[i] = (_Float16)1.0f;

  v8f acc[4] = {};
  float mrow[8], lrow[8];
#pragma unroll
  for (int r = 0; r < 8; ++r) { mrow[r] = NEGINF; lrow[r] = 0.0f; }

  const int nsteps = (qb + 64) >> 5;   // causal bound
  attn_issue_stage(Kbh, VTbh, 0, Ks[0], VTs[0], tid);

  for (int it = 0; it < nsteps; ++it) {
    const int jb = it << 5;
    __syncthreads();                   // readers of the buffer we write next done
    if (it + 1 < nsteps) {
      attn_issue_stage(Kbh, VTbh, jb + 32, Ks[(it + 1) & 1], VTs[(it + 1) & 1], tid);
      WAIT_ASYNC(4);
    } else {
      WAIT_ASYNC(0);
    }
    __syncthreads();

    const _Float16* Ksb  = Ks[it & 1];
    const _Float16* VTsb = VTs[it & 1];

    // S (16 x 32 keys) as two 16x16 C tiles
    v8f S[2];
#pragma unroll
    for (int jt = 0; jt < 2; ++jt) {
      v8f s = {};
      v16h b0 = load_B(Ksb + jt * 16 * 64, 64, 0, lane);
      v16h b1 = load_B(Ksb + jt * 16 * 64, 64, 32, lane);
      s = wmma_f16(qa0, b0, s);
      s = wmma_f16(qa1, b1, s);
      S[jt] = s;
    }

    // causal mask: only edge tiles can clip (wave-uniform branch)
    if (jb + 32 > qbase) {
#pragma unroll
      for (int jt = 0; jt < 2; ++jt) {
        int j = jb + jt * 16 + n0;
#pragma unroll
        for (int r = 0; r < 8; ++r) {
          int i = qbase + ((lane < 16) ? r : r + 8);
          if (j > i) S[jt][r] = NEGINF;
        }
      }
    }
    // key mask (general; all-true in this problem instance)
#pragma unroll
    for (int jt = 0; jt < 2; ++jt) {
      bool mok = maskb[jb + jt * 16 + n0] != 0;
#pragma unroll
      for (int r = 0; r < 8; ++r)
        if (!mok) S[jt][r] = NEGINF;
    }

    // online softmax: row max via lane shuffles, probabilities to LDS
    float corr[8];
#pragma unroll
    for (int r = 0; r < 8; ++r) {
      float mx = fmaxf(S[0][r], S[1][r]);
#pragma unroll
      for (int d = 8; d >= 1; d >>= 1) mx = fmaxf(mx, __shfl_xor(mx, d, 32));
      float mnew = fmaxf(mrow[r], mx);
      corr[r] = __expf(mrow[r] - mnew);
      mrow[r] = mnew;
      float p0 = __expf(S[0][r] - mnew);
      float p1 = __expf(S[1][r] - mnew);
#pragma unroll
      for (int t = 0; t < 4; ++t) acc[t][r] *= corr[r];
      int m = (lane < 16) ? r : r + 8;
      Ps[wid][m * 32 + n0]      = (_Float16)p0;
      Ps[wid][m * 32 + 16 + n0] = (_Float16)p1;
    }

    // same-wave LDS RAW: make P visible before A-layout reload
    asm volatile("s_wait_dscnt 0" ::: "memory");

    v16h pa = load_A(&Ps[wid][0], 32, 0, lane);

    // row sums of P via WMMA against ones: C[r] = rowsum, replicated over N
    v8f zero = {};
    v8f ssum = wmma_f16(pa, ones, zero);
#pragma unroll
    for (int r = 0; r < 8; ++r) lrow[r] = lrow[r] * corr[r] + ssum[r];

#pragma unroll
    for (int t = 0; t < 4; ++t) {
      v16h bv = load_B(VTsb + t * 16 * 32, 32, 0, lane);
      acc[t] = wmma_f16(pa, bv, acc[t]);
    }
  }

  // normalize + write (B, N, H*DH) f16
#pragma unroll
  for (int r = 0; r < 8; ++r) {
    float inv = 1.0f / lrow[r];
    int row = qbase + ((lane < 16) ? r : r + 8);
    _Float16* orow = Out + ((size_t)b * KN + row) * (KH * KDH) + h * KDH;
#pragma unroll
    for (int t = 0; t < 4; ++t)
      orow[t * 16 + n0] = (_Float16)(acc[t][r] * inv);
  }
}

// ---------------------------------------------------------------------------
// Launch
// ---------------------------------------------------------------------------
extern "C" void kernel_launch(void* const* d_in, const int* in_sizes, int n_in,
                              void* d_out, int out_size, void* d_ws, size_t ws_size,
                              hipStream_t stream) {
  const float* x            = (const float*)d_in[0];              // (B,N,D)
  const unsigned char* mask = (const unsigned char*)d_in[1];      // (B,N) bool
  const float* rope         = (const float*)d_in[2];              // (N,DH)
  const float* Wqkv         = (const float*)d_in[3];              // (3HD, D)
  const float* Wout         = (const float*)d_in[4];              // (D, HD)
  const float* bout         = (const float*)d_in[5];              // (D,)
  float* out = (float*)d_out;

  const int MT = KB * KN;            // 4096 rows
  const int NQKV = 3 * KH * KDH;     // 3072
  const int HD = KH * KDH;           // 1024

  _Float16* xh    = (_Float16*)d_ws;                 // MT*KD
  _Float16* wqkvh = xh + (size_t)MT * KD;            // NQKV*KD
  _Float16* wouth = wqkvh + (size_t)NQKV * KD;       // KD*HD
  _Float16* qkvh  = wouth + (size_t)KD * HD;         // MT*NQKV
  _Float16* qh    = qkvh + (size_t)MT * NQKV;        // B*H*N*DH
  _Float16* kh    = qh + (size_t)KB * KH * KN * KDH;
  _Float16* vth   = kh + (size_t)KB * KH * KN * KDH;
  _Float16* attnh = xh;   // reuse: x no longer needed after QKV GEMM

  int nx = MT * KD, nwq = NQKV * KD, nwo = KD * HD;
  cvt_f32_f16<<<(nx + 255) / 256, 256, 0, stream>>>(x, xh, nx);
  cvt_f32_f16<<<(nwq + 255) / 256, 256, 0, stream>>>(Wqkv, wqkvh, nwq);
  cvt_f32_f16<<<(nwo + 255) / 256, 256, 0, stream>>>(Wout, wouth, nwo);

  // qkv = x @ Wqkv^T   (4096 x 1024) * (3072 x 1024)^T -> f16
  gemm_wmma<true, false><<<dim3(MT / 64, NQKV / 256), 256, 0, stream>>>(
      xh, wqkvh, nullptr, nullptr, qkvh, MT, NQKV, KD);

  int npair = MT * NQKV / 2;
  rope_split<<<(npair + 255) / 256, 256, 0, stream>>>(qkvh, rope, qh, kh, vth);

  flash_attn<<<dim3(KN / 64, KH, KB), 128, 0, stream>>>(qh, kh, vth, mask, attnh);

  // out = attn @ Wout^T + b   (4096 x 1024) * (1024 x 1024)^T -> f32
  gemm_wmma<false, true><<<dim3(MT / 64, HD / 256), 256, 0, stream>>>(
      attnh, wouth, bout, out, nullptr, MT, HD, KD);
}
